// Decoder_26809185861705
// MI455X (gfx1250) — matrix operand order repaired and checked by reading, not compiled, not run on hardware
//
#include <hip/hip_runtime.h>
#include <hip/hip_bf16.h>
#include <stdint.h>

// ---------------------------------------------------------------------------
// Batched greedy GRU decoder for MI455X (gfx1250, wave32).
//   V=50257 vocab, E=256 embed, H=512 hidden, B=64 batch, L=64 steps.
// Classifier GEMM (64x512 @ 512x50257) done with v_wmma_f32_16x16x32_bf16;
// Wc converted once to bf16 (51.5 MB -> resident in 192MB L2).
// Hidden-state A tiles moved to LDS via the Tensor Data Mover.
// ---------------------------------------------------------------------------

#define V_SIZE 50257
#define E_SIZE 256
#define H_SIZE 512
#define B_SIZE 64
#define L_SEQ  64
#define SOS_TOK 2
#define NTILE  32                              // logit columns per block
#define NBLK   ((V_SIZE + NTILE - 1) / NTILE)  // 1571 classifier blocks

typedef __attribute__((ext_vector_type(16))) __bf16 v16bf;
typedef __attribute__((ext_vector_type(8)))  __bf16 v8bf;
typedef __attribute__((ext_vector_type(8)))  float  v8f;
typedef __attribute__((ext_vector_type(4)))  unsigned int u32x4;
typedef __attribute__((ext_vector_type(8)))  int i32x8;
typedef __attribute__((ext_vector_type(4)))  int i32x4;

#define NEG_INF (-3.402823466e38f)

__device__ __forceinline__ unsigned short f2bf(float f) {
    unsigned int u = __float_as_uint(f);
    u = (u + 0x7FFFu + ((u >> 16) & 1u)) >> 16;   // round-to-nearest-even
    return (unsigned short)u;
}

// ---------------------------------------------------------------------------
// One-time fp32 -> bf16 conversion of the classifier weight (V*H elements).
// ---------------------------------------------------------------------------
__global__ void convert_bf16_kernel(const float* __restrict__ src,
                                    unsigned short* __restrict__ dst, int n) {
    int i = blockIdx.x * blockDim.x + threadIdx.x;
    int stride = gridDim.x * blockDim.x;
    for (; i < n; i += stride) dst[i] = f2bf(src[i]);
}

// ---------------------------------------------------------------------------
// GRU cell step: x = emb[tok]; gi = x*Wih^T+bih; gh = h*Whh^T+bhh; gates.
// One block per batch row b; 256 threads handle H=512 (2 outputs each).
// Writes h_new (fp32, in place OK) and a bf16 copy for the WMMA classifier.
// ---------------------------------------------------------------------------
__global__ void __launch_bounds__(256)
gru_step_kernel(const float* __restrict__ emb,
                const float* __restrict__ Wih,   // [3H, E]
                const float* __restrict__ Whh,   // [3H, H]
                const float* __restrict__ bih,   // [3H]
                const float* __restrict__ bhh,   // [3H]
                const float* __restrict__ hprev, // [B, H]
                const int*   __restrict__ tok,   // [B]
                int use_sos,
                float* __restrict__ hout,        // [B, H]
                unsigned short* __restrict__ hbf)// [B, H] bf16
{
    __shared__ float xs[E_SIZE];
    __shared__ float hs[H_SIZE];
    const int b   = blockIdx.x;
    const int tid = threadIdx.x;

    const int t = use_sos ? SOS_TOK : tok[b];
    xs[tid] = emb[(size_t)t * E_SIZE + tid];           // tid < 256 == E
    hs[tid]       = hprev[b * H_SIZE + tid];
    hs[tid + 256] = hprev[b * H_SIZE + tid + 256];
    __syncthreads();

    #pragma unroll
    for (int jj = 0; jj < 2; ++jj) {
        const int j = tid + jj * 256;
        const float* wr = Wih + (size_t)j * E_SIZE;
        const float* wz = Wih + (size_t)(H_SIZE + j) * E_SIZE;
        const float* wn = Wih + (size_t)(2 * H_SIZE + j) * E_SIZE;
        float ir = bih[j], iz = bih[H_SIZE + j], inn = bih[2 * H_SIZE + j];
        #pragma unroll 4
        for (int k = 0; k < E_SIZE; ++k) {
            const float x = xs[k];
            ir  = fmaf(wr[k], x, ir);
            iz  = fmaf(wz[k], x, iz);
            inn = fmaf(wn[k], x, inn);
        }
        const float* ur = Whh + (size_t)j * H_SIZE;
        const float* uz = Whh + (size_t)(H_SIZE + j) * H_SIZE;
        const float* un = Whh + (size_t)(2 * H_SIZE + j) * H_SIZE;
        float hr = bhh[j], hz = bhh[H_SIZE + j], hn = bhh[2 * H_SIZE + j];
        #pragma unroll 4
        for (int k = 0; k < H_SIZE; ++k) {
            const float h = hs[k];
            hr = fmaf(ur[k], h, hr);
            hz = fmaf(uz[k], h, hz);
            hn = fmaf(un[k], h, hn);
        }
        const float r = 1.0f / (1.0f + __expf(-(ir + hr)));
        const float z = 1.0f / (1.0f + __expf(-(iz + hz)));
        const float n = tanhf(inn + r * hn);
        const float hv = (1.0f - z) * n + z * hs[j];
        hout[b * H_SIZE + j] = hv;
        hbf [b * H_SIZE + j] = f2bf(hv);
    }
}

// ---------------------------------------------------------------------------
// Classifier: logits[b, v0..v0+31] = h_bf[64,512] @ Wc_bf[v,:]^T + bc.
// 8 waves/block: wave w -> M-tile (w&3), N-subtile (w>>2).
// A (hidden state) DMA'd into LDS by the Tensor Data Mover in two 64x256
// bf16 slabs; per-slab 8 unrolled K=32 WMMA steps -> 16 wmma total.
// Block also emits a per-row argmax candidate over its 32 columns.
// ---------------------------------------------------------------------------
__global__ void __launch_bounds__(256)
classify_step_kernel(const unsigned short* __restrict__ hbf,   // [64,512] bf16
                     const unsigned short* __restrict__ wcbf,  // [V,512] bf16
                     const float* __restrict__ bc,             // [V]
                     float* __restrict__ out,                  // logits [64,V]
                     float* __restrict__ candVal,              // [NBLK,64]
                     int*   __restrict__ candIdx)              // [NBLK,64]
{
    __shared__ __attribute__((aligned(32))) unsigned short ldsA[B_SIZE * 256]; // 32KB slab
    __shared__ float cVal[B_SIZE * 2];
    __shared__ int   cIdx[B_SIZE * 2];

    const int tid  = threadIdx.x;
    const int lane = tid & 31;
    const int wave = tid >> 5;
    const int half = lane >> 4;
    const int l16  = lane & 15;
    const int m    = wave & 3;                      // M tile (rows m*16..m*16+15)
    const int nsub = wave >> 2;                     // N sub-tile within block
    const int v0   = blockIdx.x * NTILE + nsub * 16;
    const int Ng   = v0 + l16;                      // this lane's logit column
    const int Ngc  = (Ng < V_SIZE) ? Ng : (V_SIZE - 1);
    const int R    = m * 16 + l16;                  // this lane's A row

    v8f acc = {0.f, 0.f, 0.f, 0.f, 0.f, 0.f, 0.f, 0.f};

    for (int p = 0; p < 2; ++p) {                   // two K-halves of 256
        if (tid < 32) {
            // Tensor DMA: 64 x 256 bf16 tile of hbf -> LDS.
            const unsigned int ldsAddr = (unsigned int)(size_t)(void*)ldsA;
            const uint64_t ga = (uint64_t)(uintptr_t)hbf + (uint64_t)p * 256u * 2u;
            u32x4 g0 = { 1u,                                   // count=1, user D#
                         ldsAddr,                              // lds_addr
                         (unsigned int)ga,                     // global_addr lo
                         (unsigned int)((ga >> 32) & 0x1FFFFFFu) | 0x80000000u }; // hi + type=2
            i32x8 g1 = { 0x00010000,          // data_size=2B, no mask/pad/iter
                         (int)(512u << 16),   // tensor_dim0 = 512
                         (int)(64u  << 16),   // tensor_dim1 = 64
                         (int)(256u << 16),   // tile_dim0   = 256
                         64,                  // tile_dim1   = 64, tile_dim2 = 0
                         512,                 // tensor_dim0_stride = 512
                         0, 0 };
            i32x4 g2 = {0, 0, 0, 0};
            i32x4 g3 = {0, 0, 0, 0};
            i32x8 g4 = {0, 0, 0, 0, 0, 0, 0, 0};
            __builtin_amdgcn_tensor_load_to_lds(g0, g1, g2, g3, g4, 0);
            __builtin_amdgcn_s_wait_tensorcnt(0);
        }
        __syncthreads();

        #pragma unroll
        for (int kc = 0; kc < 8; ++kc) {
            // A fragment: elements 0..15 -> K=16*half+e (contiguous, matches B)
            // within this 32-wide K chunk.
            const int a0off = R * 256 + kc * 32 + 16 * half;
            v8bf a0 = *(const v8bf*)&ldsA[a0off];
            v8bf a1 = *(const v8bf*)&ldsA[a0off + 8];
            v16bf a = __builtin_shufflevector(a0, a1,
                        0, 1, 2, 3, 4, 5, 6, 7, 8, 9, 10, 11, 12, 13, 14, 15);
            // B fragment: lane holds 16 contiguous K for column Ng
            // (K = kchunk + 16*half + e), straight from row-major Wc_bf.
            const size_t boff = (size_t)Ngc * H_SIZE + (size_t)p * 256 + kc * 32 + 16 * half;
            v16bf bfrag = *(const v16bf*)&wcbf[boff];

            acc = __builtin_amdgcn_wmma_f32_16x16x32_bf16(
                      false, a, false, bfrag, (short)0, acc, false, false);
        }
        __syncthreads();
    }

    // Bias, store logits, per-lane values for argmax.
    const float bias = bc[Ngc];
    const bool  inb  = (Ng < V_SIZE);
    float bvals[8];
    int   bidx[8];
    #pragma unroll
    for (int i = 0; i < 8; ++i) {
        const int row = m * 16 + i + 8 * half;      // D layout: VGPR i, half selects +8
        const float v = acc[i] + bias;
        if (inb) out[(size_t)row * V_SIZE + Ng] = v;
        bvals[i] = inb ? v : NEG_INF;
        bidx[i]  = Ng;
    }
    // Butterfly argmax across the 16 lanes of each half (ties -> lowest index).
    #pragma unroll
    for (int off = 1; off < 16; off <<= 1) {
        #pragma unroll
        for (int i = 0; i < 8; ++i) {
            const float ov = __shfl_xor(bvals[i], off, 32);
            const int   oi = __shfl_xor(bidx[i],  off, 32);
            if (ov > bvals[i] || (ov == bvals[i] && oi < bidx[i])) {
                bvals[i] = ov;
                bidx[i]  = oi;
            }
        }
    }
    if (l16 == 0) {
        #pragma unroll
        for (int i = 0; i < 8; ++i) {
            const int row = m * 16 + i + 8 * half;
            cVal[row * 2 + nsub] = bvals[i];
            cIdx[row * 2 + nsub] = bidx[i];
        }
    }
    __syncthreads();
    if (tid < B_SIZE) {
        const float va = cVal[tid * 2], vb = cVal[tid * 2 + 1];
        const int   ia = cIdx[tid * 2], ib = cIdx[tid * 2 + 1];
        const bool takeB = (vb > va) || (vb == va && ib < ia);
        candVal[(size_t)blockIdx.x * B_SIZE + tid] = takeB ? vb : va;
        candIdx[(size_t)blockIdx.x * B_SIZE + tid] = takeB ? ib : ia;
    }
}

// ---------------------------------------------------------------------------
// Reduce per-block argmax candidates to the next token per batch row.
// ---------------------------------------------------------------------------
__global__ void __launch_bounds__(256)
argmax_reduce_kernel(const float* __restrict__ candVal,
                     const int*   __restrict__ candIdx,
                     int* __restrict__ tok)
{
    __shared__ float sv[256];
    __shared__ int   si[256];
    const int b = blockIdx.x, tid = threadIdx.x;
    float bv = NEG_INF;
    int   bi = 0x7fffffff;
    for (int i = tid; i < NBLK; i += 256) {
        const float v = candVal[(size_t)i * B_SIZE + b];
        const int  ix = candIdx[(size_t)i * B_SIZE + b];
        if (v > bv || (v == bv && ix < bi)) { bv = v; bi = ix; }
    }
    sv[tid] = bv; si[tid] = bi;
    __syncthreads();
    for (int s = 128; s > 0; s >>= 1) {
        if (tid < s) {
            const float ov = sv[tid + s];
            const int   oi = si[tid + s];
            if (ov > sv[tid] || (ov == sv[tid] && oi < si[tid])) {
                sv[tid] = ov; si[tid] = oi;
            }
        }
        __syncthreads();
    }
    if (tid == 0) tok[b] = si[0];
}

// ---------------------------------------------------------------------------
extern "C" void kernel_launch(void* const* d_in, const int* in_sizes, int n_in,
                              void* d_out, int out_size, void* d_ws, size_t ws_size,
                              hipStream_t stream)
{
    const float* h0  = (const float*)d_in[0];   // [1,B,H]
    const float* emb = (const float*)d_in[1];   // [V,E]
    const float* Wih = (const float*)d_in[2];   // [3H,E]
    const float* Whh = (const float*)d_in[3];   // [3H,H]
    const float* bih = (const float*)d_in[4];   // [3H]
    const float* bhh = (const float*)d_in[5];   // [3H]
    const float* Wc  = (const float*)d_in[6];   // [V,H]
    const float* bc  = (const float*)d_in[7];   // [V]
    float* out = (float*)d_out;                 // [L,B,V]

    char* ws = (char*)d_ws;
    size_t off = 0;
    auto alloc = [&](size_t bytes) {
        void* p = ws + off;
        off = (off + bytes + 255) & ~(size_t)255;
        return p;
    };
    unsigned short* wcbf = (unsigned short*)alloc((size_t)V_SIZE * H_SIZE * 2);
    float*          hbuf = (float*)         alloc((size_t)B_SIZE * H_SIZE * 4);
    unsigned short* hbf  = (unsigned short*)alloc((size_t)B_SIZE * H_SIZE * 2);
    int*            tok  = (int*)           alloc((size_t)B_SIZE * 4);
    float*       candVal = (float*)         alloc((size_t)NBLK * B_SIZE * 4);
    int*         candIdx = (int*)           alloc((size_t)NBLK * B_SIZE * 4);
    (void)ws_size; (void)in_sizes; (void)n_in; (void)out_size;

    // One-time: Wc fp32 -> bf16 (51.5 MB, L2-resident thereafter).
    convert_bf16_kernel<<<2048, 256, 0, stream>>>(Wc, wcbf, V_SIZE * H_SIZE);

    for (int step = 0; step < L_SEQ; ++step) {
        const float* hprev = (step == 0) ? h0 : hbuf;
        gru_step_kernel<<<B_SIZE, 256, 0, stream>>>(
            emb, Wih, Whh, bih, bhh, hprev, tok, step == 0 ? 1 : 0, hbuf, hbf);
        classify_step_kernel<<<NBLK, 256, 0, stream>>>(
            hbf, wcbf, bc, out + (size_t)step * B_SIZE * V_SIZE, candVal, candIdx);
        if (step + 1 < L_SEQ)
            argmax_reduce_kernel<<<B_SIZE, 256, 0, stream>>>(candVal, candIdx, tok);
    }
}
